// SparseMoEFFN_72103910965741
// MI455X (gfx1250) — matrix operand order repaired and checked by reading, not compiled, not run on hardware
//
#include <hip/hip_runtime.h>
#include <math.h>

// ---------------------------------------------------------------------------
// SparseMoEFFN for gfx1250 (MI455X): bf16 WMMA, per-expert token gathering.
// M-tile = 64 tokens, 512 threads (16 waves), B-fragments reused across
// 4 M-subtiles (64 FLOP per L2 byte), async global->LDS x staging.
// ---------------------------------------------------------------------------

typedef __attribute__((ext_vector_type(16))) __bf16 v16bf;
typedef __attribute__((ext_vector_type(8)))  __bf16 v8bf;
typedef __attribute__((ext_vector_type(8)))  float  v8f;

#define T_TOK  32768          // 8 * 4096 tokens
#define DM     512            // d_model
#define DF     2048           // d_ff
#define NE     8              // experts
#define MT     64             // tokens per block tile
#define MS     4              // M-subtiles (MT/16)
#define XPITCH 520            // LDS row pitch in bf16 elems (+4-bank shift/row)

__device__ __forceinline__ unsigned short f2bf(float f) {
    union { float f; unsigned u; } v; v.f = f;
    unsigned r = v.u + 0x7FFFu + ((v.u >> 16) & 1u);   // round-to-nearest-even
    return (unsigned short)(r >> 16);
}

__device__ __forceinline__ v16bf cat16(v8bf lo, v8bf hi) {
    return __builtin_shufflevector(lo, hi, 0,1,2,3,4,5,6,7,8,9,10,11,12,13,14,15);
}

// Async 16B copy: global memory -> LDS (GLOBAL_LOAD_ASYNC_TO_LDS_B128,
// tracked by ASYNCcnt).  lds_off = low 32 bits of the flat LDS pointer
// (LDS aperture maps addr[31:0] to the wave's LDS offset).
__device__ __forceinline__ void async_copy16(unsigned lds_off, const void* gptr) {
    unsigned long long ga = (unsigned long long)(uintptr_t)gptr;
    asm volatile("global_load_async_to_lds_b128 %0, %1, off"
                 :: "v"(lds_off), "v"(ga) : "memory");
}
__device__ __forceinline__ void wait_async0() {
    asm volatile("s_wait_asynccnt 0x0" ::: "memory");
}

// ---------------------------------------------------------------------------
// Prep kernels
// ---------------------------------------------------------------------------
__global__ void zero_out_kernel(float* __restrict__ out, int n, int* __restrict__ cnt) {
    int gid = blockIdx.x * blockDim.x + threadIdx.x;
    if (gid < NE) cnt[gid] = 0;
    for (int i = gid; i < n; i += gridDim.x * blockDim.x) out[i] = 0.0f;
}

__global__ void cvt_x_kernel(const float* __restrict__ x, unsigned short* __restrict__ xb, int n) {
    int gid = blockIdx.x * blockDim.x + threadIdx.x;
    for (int i = gid; i < n; i += gridDim.x * blockDim.x) xb[i] = f2bf(x[i]);
}

// w1[e][k][n] (k<DM, n<DF)  ->  w1t[e][n][k]  (row n contiguous in K=DM)
__global__ void tr_w1_kernel(const float* __restrict__ w1, unsigned short* __restrict__ w1t) {
    int gid = blockIdx.x * blockDim.x + threadIdx.x;
    const int n_tot = NE * DM * DF;
    for (int i = gid; i < n_tot; i += gridDim.x * blockDim.x) {
        int e = i / (DM * DF);
        int r = i % (DM * DF);
        int k = r / DF;
        int n = r % DF;
        w1t[((size_t)e * DF + n) * DM + k] = f2bf(w1[i]);
    }
}

// w2[e][k][n] (k<DF, n<DM)  ->  w2t[e][n][k]  (row n contiguous in K=DF)
__global__ void tr_w2_kernel(const float* __restrict__ w2, unsigned short* __restrict__ w2t) {
    int gid = blockIdx.x * blockDim.x + threadIdx.x;
    const int n_tot = NE * DM * DF;
    for (int i = gid; i < n_tot; i += gridDim.x * blockDim.x) {
        int e = i / (DF * DM);
        int r = i % (DF * DM);
        int k = r / DM;
        int n = r % DM;
        w2t[((size_t)e * DM + n) * DF + k] = f2bf(w2[i]);
    }
}

// ---------------------------------------------------------------------------
// Router: logits -> softmax -> top-2 -> normalized weights -> gather lists
// ---------------------------------------------------------------------------
__global__ __launch_bounds__(256) void moe_router_kernel(
    const float* __restrict__ x, const float* __restrict__ rw,
    const float* __restrict__ rb, int* __restrict__ cnt,
    int* __restrict__ toks, float* __restrict__ wts)
{
    __shared__ float rws[DM * NE];                       // 16 KB
    int tid = threadIdx.x;
    for (int i = tid; i < DM * NE; i += 256) rws[i] = rw[i];
    __syncthreads();

    int t = blockIdx.x * 256 + tid;
    float acc[NE];
    #pragma unroll
    for (int e = 0; e < NE; ++e) acc[e] = rb[e];
    const float* xr = x + (size_t)t * DM;
    for (int d = 0; d < DM; ++d) {
        float xv = xr[d];
        #pragma unroll
        for (int e = 0; e < NE; ++e) acc[e] = fmaf(xv, rws[d * NE + e], acc[e]);
    }
    // full softmax (matches reference numerics)
    float mx = acc[0];
    #pragma unroll
    for (int e = 1; e < NE; ++e) mx = fmaxf(mx, acc[e]);
    float p[NE]; float s = 0.0f;
    #pragma unroll
    for (int e = 0; e < NE; ++e) { p[e] = __expf(acc[e] - mx); s += p[e]; }
    float inv = 1.0f / s;
    #pragma unroll
    for (int e = 0; e < NE; ++e) p[e] *= inv;
    // top-2 (first index wins ties, like jax top_k)
    int i1 = 0; float p1 = p[0];
    #pragma unroll
    for (int e = 1; e < NE; ++e) if (p[e] > p1) { p1 = p[e]; i1 = e; }
    int i2 = (i1 == 0) ? 1 : 0; float p2 = p[i2];
    #pragma unroll
    for (int e = 0; e < NE; ++e) if (e != i1 && p[e] > p2) { p2 = p[e]; i2 = e; }
    float denom = p1 + p2 + 1e-9f;
    float wA = p1 / denom, wB = p2 / denom;

    int pos1 = atomicAdd(&cnt[i1], 1);
    toks[i1 * T_TOK + pos1] = t;  wts[i1 * T_TOK + pos1] = wA;
    int pos2 = atomicAdd(&cnt[i2], 1);
    toks[i2 * T_TOK + pos2] = t;  wts[i2 * T_TOK + pos2] = wB;
}

// ---------------------------------------------------------------------------
// Fused expert FFN: GEMM1 (64xDM @ DMxDF) -> GELU -> GEMM2 (64xDF @ DFxDM)
// 512 threads = 16 waves; d_ff processed in 4 chunks of 512.
// Per wave: 2 N-tiles x 4 M-subtiles; each B fragment feeds 4 WMMAs.
// ---------------------------------------------------------------------------
__global__ __launch_bounds__(512) void moe_ffn_kernel(
    const unsigned short* __restrict__ xb,
    const unsigned short* __restrict__ w1t,
    const unsigned short* __restrict__ w2t,
    const float* __restrict__ b1, const float* __restrict__ b2,
    const int* __restrict__ cnt, const int* __restrict__ toks,
    const float* __restrict__ wts, float* __restrict__ out)
{
    int e = blockIdx.y;
    int ntok = cnt[e];
    int base = blockIdx.x * MT;
    if (base >= ntok) return;

    extern __shared__ __align__(16) unsigned short smem[];
    unsigned short* x_lds = smem;                  // MT * XPITCH bf16 (66.5 KB)
    unsigned short* h_lds = smem + MT * XPITCH;    // MT * XPITCH bf16 (66.5 KB)
    __shared__ int   tok_s[MT];
    __shared__ float wt_s[MT];

    int tid = threadIdx.x;
    if (tid < MT) {
        int idx = base + tid;
        tok_s[tid] = (idx < ntok) ? toks[e * T_TOK + idx] : toks[e * T_TOK];
        wt_s[tid]  = (idx < ntok) ? wts[e * T_TOK + idx]  : 0.0f;
    }
    __syncthreads();

    // stage x tile (MT rows x 512 bf16) via async global->LDS 16B copies
    for (int chunk = tid; chunk < MT * 64; chunk += 512) {
        int row = chunk >> 6, c8 = chunk & 63;
        unsigned lds_off = (unsigned)(uintptr_t)&x_lds[row * XPITCH + c8 * 8];
        async_copy16(lds_off, &xb[(size_t)tok_s[row] * DM + c8 * 8]);
    }
    wait_async0();
    __syncthreads();

    int lane = tid & 31, wave = tid >> 5;   // 16 waves
    int m  = lane & 15;      // fragment row (A) / column (B,D)
    int hi = lane >> 4;      // half-wave selector

    v8f zacc = {};
    v8f acc2[2][MS];         // [n-tile][m-subtile] persist over chunks
    #pragma unroll
    for (int t2 = 0; t2 < 2; ++t2)
        #pragma unroll
        for (int ms = 0; ms < MS; ++ms) acc2[t2][ms] = zacc;

    for (int c = 0; c < 4; ++c) {
        // ---------------- GEMM1 for this d_ff chunk ----------------
        v8f acc1[2][MS];
        #pragma unroll
        for (int t2 = 0; t2 < 2; ++t2)
            #pragma unroll
            for (int ms = 0; ms < MS; ++ms) acc1[t2][ms] = zacc;

        const unsigned short* w1base[2];
        #pragma unroll
        for (int t2 = 0; t2 < 2; ++t2) {
            int nglob = c * 512 + wave * 32 + t2 * 16 + m;
            w1base[t2] = w1t + ((size_t)e * DF + nglob) * DM;
        }
        for (int kb = 0; kb < DM; kb += 32) {
            v16bf a[MS];
            #pragma unroll
            for (int ms = 0; ms < MS; ++ms) {
                int row = ms * 16 + m;
                v8bf alo = *(const v8bf*)&x_lds[row * XPITCH + kb + hi * 8];
                v8bf ahi = *(const v8bf*)&x_lds[row * XPITCH + kb + 16 + hi * 8];
                a[ms] = cat16(alo, ahi);
            }
            #pragma unroll
            for (int t2 = 0; t2 < 2; ++t2) {
                const unsigned short* bp = w1base[t2] + kb + hi * 8;
                v16bf b = cat16(*(const v8bf*)bp, *(const v8bf*)(bp + 16));
                #pragma unroll
                for (int ms = 0; ms < MS; ++ms)
                    acc1[t2][ms] = __builtin_amdgcn_wmma_f32_16x16x32_bf16(
                        false, a[ms], false, b, (short)0, acc1[t2][ms], false, false);
            }
        }

        // WAR guard: previous chunk's GEMM2 reads of h_lds must be done
        __syncthreads();

        // GELU epilogue -> h chunk (bf16) in LDS
        #pragma unroll
        for (int t2 = 0; t2 < 2; ++t2) {
            int nloc  = wave * 32 + t2 * 16 + m;
            float bias = b1[e * DF + c * 512 + nloc];
            #pragma unroll
            for (int ms = 0; ms < MS; ++ms) {
                #pragma unroll
                for (int r = 0; r < 8; ++r) {
                    float v = acc1[t2][ms][r] + bias;
                    float g = 0.5f * v * (1.0f + erff(v * 0.70710678118f));
                    h_lds[(ms * 16 + r + 8 * hi) * XPITCH + nloc] = f2bf(g);
                }
            }
        }
        __syncthreads();

        // ---------------- GEMM2 partial over this K chunk ----------------
        const unsigned short* w2base[2];
        #pragma unroll
        for (int t2 = 0; t2 < 2; ++t2) {
            int n2 = wave * 32 + t2 * 16 + m;
            w2base[t2] = w2t + ((size_t)e * DM + n2) * DF + c * 512;
        }
        for (int kb = 0; kb < 512; kb += 32) {
            v16bf a[MS];
            #pragma unroll
            for (int ms = 0; ms < MS; ++ms) {
                int row = ms * 16 + m;
                v8bf alo = *(const v8bf*)&h_lds[row * XPITCH + kb + hi * 8];
                v8bf ahi = *(const v8bf*)&h_lds[row * XPITCH + kb + 16 + hi * 8];
                a[ms] = cat16(alo, ahi);
            }
            #pragma unroll
            for (int t2 = 0; t2 < 2; ++t2) {
                const unsigned short* bp = w2base[t2] + kb + hi * 8;
                v16bf b = cat16(*(const v8bf*)bp, *(const v8bf*)(bp + 16));
                #pragma unroll
                for (int ms = 0; ms < MS; ++ms)
                    acc2[t2][ms] = __builtin_amdgcn_wmma_f32_16x16x32_bf16(
                        false, a[ms], false, b, (short)0, acc2[t2][ms], false, false);
            }
        }
    }

    // ---------------- output: + b2, * combine weight, scatter-add ----------
    #pragma unroll
    for (int t2 = 0; t2 < 2; ++t2) {
        int n2 = wave * 32 + t2 * 16 + m;
        float bias2 = b2[e * DM + n2];
        #pragma unroll
        for (int ms = 0; ms < MS; ++ms) {
            #pragma unroll
            for (int r = 0; r < 8; ++r) {
                int mr = ms * 16 + r + 8 * hi;
                float w = wt_s[mr];
                if (w != 0.0f)
                    atomicAdd(&out[(size_t)tok_s[mr] * DM + n2],
                              w * (acc2[t2][ms][r] + bias2));
            }
        }
    }
}

// ---------------------------------------------------------------------------
// Launch
// ---------------------------------------------------------------------------
extern "C" void kernel_launch(void* const* d_in, const int* in_sizes, int n_in,
                              void* d_out, int out_size, void* d_ws, size_t ws_size,
                              hipStream_t stream) {
    const float* x  = (const float*)d_in[0];
    const float* rw = (const float*)d_in[1];
    const float* rb = (const float*)d_in[2];
    const float* w1 = (const float*)d_in[3];
    const float* b1 = (const float*)d_in[4];
    const float* w2 = (const float*)d_in[5];
    const float* b2 = (const float*)d_in[6];
    float* out = (float*)d_out;

    char* ws = (char*)d_ws;
    size_t off = 0;
    unsigned short* xb  = (unsigned short*)(ws + off); off += (size_t)T_TOK * DM * 2;   // 33.5 MB
    unsigned short* w1t = (unsigned short*)(ws + off); off += (size_t)NE * DM * DF * 2; // 16.8 MB
    unsigned short* w2t = (unsigned short*)(ws + off); off += (size_t)NE * DM * DF * 2; // 16.8 MB
    int*   cnt  = (int*)(ws + off);   off += 256;
    int*   toks = (int*)(ws + off);   off += (size_t)NE * T_TOK * 4;                    // 1 MB
    float* wts  = (float*)(ws + off); off += (size_t)NE * T_TOK * 4;                    // 1 MB

    zero_out_kernel<<<1024, 256, 0, stream>>>(out, T_TOK * DM, cnt);
    cvt_x_kernel  <<<1024, 256, 0, stream>>>(x, xb, T_TOK * DM);
    tr_w1_kernel  <<<2048, 256, 0, stream>>>(w1, w1t);
    tr_w2_kernel  <<<2048, 256, 0, stream>>>(w2, w2t);
    moe_router_kernel<<<T_TOK / 256, 256, 0, stream>>>(x, rw, rb, cnt, toks, wts);

    size_t smem_bytes = (size_t)2 * MT * XPITCH * sizeof(unsigned short);  // 133 KB
    dim3 g(T_TOK / MT, NE);
    moe_ffn_kernel<<<g, 512, smem_bytes, stream>>>(xb, w1t, w2t, b1, b2, cnt, toks, wts, out);
}
